// SpatialGAT_2843268350790
// MI455X (gfx1250) — compile-verified, matrix operand order checked by reading, not run on hardware
//
#include <hip/hip_runtime.h>
#include <hip/hip_bf16.h>

// ---------------------------------------------------------------------------
// 2-layer GAT for MI455X (gfx1250, wave32).
// GEMMs via V_WMMA (f32 16x16x4 preferred; bf16 16x16x32 fallback).
// Edge softmax/aggregation via L2-resident f32 atomics (all node arrays fit
// in the 192MB L2).
// ---------------------------------------------------------------------------

typedef __attribute__((ext_vector_type(2)))  float  v2f;
typedef __attribute__((ext_vector_type(8)))  float  v8f;

#define NEG_SLOPE 0.2f

#if __has_builtin(__builtin_amdgcn_wmma_f32_16x16x4_f32)
#define USE_F32_WMMA 1
#else
#define USE_F32_WMMA 0
typedef __attribute__((ext_vector_type(16))) __bf16 v16bf;
#endif

// ---------------------------------------------------------------------------
// utility: grid-stride fill
// ---------------------------------------------------------------------------
__global__ void fill_k(float* __restrict__ p, float v, size_t n) {
  size_t i = (size_t)blockIdx.x * blockDim.x + threadIdx.x;
  size_t stride = (size_t)gridDim.x * blockDim.x;
  for (; i < n; i += stride) p[i] = v;
}

// ---------------------------------------------------------------------------
// WMMA GEMM: D[M,NC] = A[M,K] * B[K,NC].  One wave per 16x16 output tile.
// K is a compile-time constant (128 here) so the WMMA chain fully unrolls.
// ---------------------------------------------------------------------------
template <int K>
__global__ __launch_bounds__(32)
void gemm_wmma_k(const float* __restrict__ A, const float* __restrict__ B,
                 float* __restrict__ D, int M, int NC) {
  const int tm   = blockIdx.x * 16;     // row-tile base
  const int tn   = blockIdx.y * 16;     // col-tile base
  const int lane = threadIdx.x & 31;
  const int half = lane >> 4;           // 0 or 1 (lane group)
  const int ml   = lane & 15;

  int mrow = tm + ml;
  if (mrow >= M) mrow = M - 1;          // clamp loads; stores are guarded
  const float* __restrict__ arow = A + (size_t)mrow * K;
  const int col = tn + ml;

  v8f acc = {};

#if USE_F32_WMMA
  // A 16x4 f32: lanes 0-15 M=0..15 hold K=0 (v0), K=1 (v1);
  //             lanes 16-31 hold K=2 (v0), K=3 (v1). B mirrors along K.
  const int kh = half * 2;
#pragma unroll
  for (int k = 0; k < K; k += 4) {
    v2f a, b;
    a.x = arow[k + kh];
    a.y = arow[k + kh + 1];
    b.x = B[(size_t)(k + kh) * NC + col];
    b.y = B[(size_t)(k + kh + 1) * NC + col];
    acc = __builtin_amdgcn_wmma_f32_16x16x4_f32(
        false, a, false, b, (short)0, acc, false, false);
  }
#else
  // bf16 16x16x32 fallback (codegen-confirmed builtin).
  // A 16x32: lane<16 holds K 0..7 (v0-3) and 16..23 (v4-7); lane>=16 holds
  // K 8..15 and 24..31.  element i of v16bf -> K = half*8 + i (i<8),
  // element 8+i -> K = 16 + half*8 + i.  B mirrors along K.
#pragma unroll
  for (int k = 0; k < K; k += 32) {
    v16bf a, b;
#pragma unroll
    for (int i = 0; i < 8; ++i) {
      const int k0 = k + half * 8 + i;
      const int k1 = k + 16 + half * 8 + i;
      a[i]     = (__bf16)arow[k0];
      a[8 + i] = (__bf16)arow[k1];
      b[i]     = (__bf16)B[(size_t)k0 * NC + col];
      b[8 + i] = (__bf16)B[(size_t)k1 * NC + col];
    }
    acc = __builtin_amdgcn_wmma_f32_16x16x32_bf16(
        false, a, false, b, (short)0, acc, false, false);
  }
#endif

  // C/D 16x16 f32 layout: VGPR r -> row M = r + half*8, lanes(&15) -> N.
#pragma unroll
  for (int r = 0; r < 8; ++r) {
    const int m = tm + half * 8 + r;
    if (m < M) D[(size_t)m * NC + tn + ml] = acc[r];
  }
}

// ---------------------------------------------------------------------------
// attention scores: a_src[n,h] = sum_c h[n,h,c]*att_src[h,c]; same for dst
// ---------------------------------------------------------------------------
template <int H>
__global__ void att_scores_k(const float* __restrict__ h,
                             const float* __restrict__ att_s,
                             const float* __restrict__ att_d,
                             float* __restrict__ as, float* __restrict__ ad,
                             int N) {
  const long idx = (long)blockIdx.x * blockDim.x + threadIdx.x;
  if (idx >= (long)N * H) return;
  const long n = idx / H;
  const int hh = (int)(idx % H);
  const float* __restrict__ hp = h + ((size_t)n * H + hh) * 32;
  float s = 0.f, d = 0.f;
#pragma unroll
  for (int c = 0; c < 32; ++c) {
    const float v = hp[c];
    s += v * att_s[hh * 32 + c];
    d += v * att_d[hh * 32 + c];
  }
  as[idx] = s;
  ad[idx] = d;
}

// ---------------------------------------------------------------------------
// float atomic max via int CAS (safe for any sign, no NaNs here)
// ---------------------------------------------------------------------------
__device__ inline void atomicMaxF(float* addr, float val) {
  unsigned int* ai = (unsigned int*)addr;
  unsigned int cur = __float_as_uint(*addr);
  while (__uint_as_float(cur) < val) {
    unsigned int old = atomicCAS(ai, cur, __float_as_uint(val));
    if (old == cur) break;
    cur = old;
  }
}

__device__ inline void edge_endpoints(const int* __restrict__ ei, long e,
                                      long E, int& s, int& d) {
  if (e < E) { s = ei[e]; d = ei[E + e]; }    // edge_index[0][e], [1][e]
  else       { s = d = (int)(e - E); }        // self loop
}

// ---------------------------------------------------------------------------
// edge pass 1: segment max of leaky_relu(a_src[src]+a_dst[dst]) over dst
// ---------------------------------------------------------------------------
template <int H>
__global__ void edge_max_k(const int* __restrict__ ei, long E, int N,
                           const float* __restrict__ as,
                           const float* __restrict__ ad,
                           float* __restrict__ amax) {
  const long idx = (long)blockIdx.x * blockDim.x + threadIdx.x;
  const long tot = (E + N) * H;
  if (idx >= tot) return;
  const long e = idx / H;
  const int hh = (int)(idx % H);
  int s, d;
  edge_endpoints(ei, e, E, s, d);
  float v = as[(long)s * H + hh] + ad[(long)d * H + hh];
  v = v > 0.f ? v : NEG_SLOPE * v;
  atomicMaxF(amax + (long)d * H + hh, v);
}

// ---------------------------------------------------------------------------
// edge pass 2: aexp[e,h] = exp(lrelu - amax[dst,h]); denom[dst,h] += aexp
// ---------------------------------------------------------------------------
template <int H>
__global__ void edge_exp_k(const int* __restrict__ ei, long E, int N,
                           const float* __restrict__ as,
                           const float* __restrict__ ad,
                           const float* __restrict__ amax,
                           float* __restrict__ aexp,
                           float* __restrict__ denom) {
  const long idx = (long)blockIdx.x * blockDim.x + threadIdx.x;
  const long tot = (E + N) * H;
  if (idx >= tot) return;
  const long e = idx / H;
  const int hh = (int)(idx % H);
  int s, d;
  edge_endpoints(ei, e, E, s, d);
  float v = as[(long)s * H + hh] + ad[(long)d * H + hh];
  v = v > 0.f ? v : NEG_SLOPE * v;
  const float ex = expf(v - amax[(long)d * H + hh]);
  aexp[idx] = ex;
  atomicAdd(denom + (long)d * H + hh, ex);
}

// ---------------------------------------------------------------------------
// edge pass 3 (layer1, H=4, F=128): agg[dst] += h[src] * aexp[e,head]
// one wave per edge, float4 per lane (16B aligned loads, 4 f32 atomics)
// ---------------------------------------------------------------------------
__global__ __launch_bounds__(256)
void edge_agg_h4_k(const int* __restrict__ ei, long E, int N,
                   const float* __restrict__ h,
                   const float* __restrict__ aexp,
                   float* __restrict__ agg) {
  const long e = (long)blockIdx.x * 8 + (threadIdx.x >> 5);
  if (e >= E + N) return;
  const int lane = threadIdx.x & 31;
  int s, d;
  edge_endpoints(ei, e, E, s, d);
  const int c4 = lane * 4;        // 0..124
  const int head = lane >> 3;     // c4 / 32
  const float al = aexp[e * 4 + head];
  const float4 hv = *(const float4*)(h + (size_t)s * 128 + c4);
  float* o = agg + (size_t)d * 128 + c4;
  atomicAdd(o + 0, hv.x * al);
  atomicAdd(o + 1, hv.y * al);
  atomicAdd(o + 2, hv.z * al);
  atomicAdd(o + 3, hv.w * al);
}

// edge pass 3 (layer2, H=1, C=32): one wave per edge, one float per lane
__global__ __launch_bounds__(256)
void edge_agg_h1_k(const int* __restrict__ ei, long E, int N,
                   const float* __restrict__ h,
                   const float* __restrict__ aexp,
                   float* __restrict__ agg) {
  const long e = (long)blockIdx.x * 8 + (threadIdx.x >> 5);
  if (e >= E + N) return;
  const int c = threadIdx.x & 31;
  int s, d;
  edge_endpoints(ei, e, E, s, d);
  const float al = aexp[e];
  atomicAdd(agg + (size_t)d * 32 + c, h[(size_t)s * 32 + c] * al);
}

// ---------------------------------------------------------------------------
// epilogue: out[n,f] = relu(agg[n,f] / (denom[n,h]+1e-16) + bias[f])
// (node-level normalization == per-edge alpha normalization, exactly)
// ---------------------------------------------------------------------------
__global__ void epilogue_k(const float* __restrict__ agg,
                           const float* __restrict__ denom,
                           const float* __restrict__ bias,
                           float* __restrict__ out, int N, int H, int C) {
  const long idx = (long)blockIdx.x * blockDim.x + threadIdx.x;
  const int F = H * C;
  if (idx >= (long)N * F) return;
  const long n = idx / F;
  const int f = (int)(idx % F);
  const int hh = f / C;
  float v = agg[idx] / (denom[n * H + hh] + 1e-16f) + bias[f];
  out[idx] = v > 0.f ? v : 0.f;
}

// ---------------------------------------------------------------------------
extern "C" void kernel_launch(void* const* d_in, const int* in_sizes, int n_in,
                              void* d_out, int out_size, void* d_ws, size_t ws_size,
                              hipStream_t stream) {
  const float* x        = (const float*)d_in[0];
  const int*   ei       = (const int*)  d_in[1];
  const float* W1       = (const float*)d_in[2];
  const float* att_src1 = (const float*)d_in[3];
  const float* att_dst1 = (const float*)d_in[4];
  const float* bias1    = (const float*)d_in[5];
  const float* W2       = (const float*)d_in[6];
  const float* att_src2 = (const float*)d_in[7];
  const float* att_dst2 = (const float*)d_in[8];
  const float* bias2    = (const float*)d_in[9];
  float* out = (float*)d_out;

  const int  N  = in_sizes[0] / 128;   // IN_C = 128
  const long E  = in_sizes[1] / 2;     // edge_index is [2,E]
  const long Et = E + N;               // edges + self loops

  // workspace layout (floats)
  float* ws = (float*)d_ws;
  size_t off = 0;
  float* h1    = ws + off; off += (size_t)N * 128;   // also reused as h2 base region? no: h2 aliases h1
  float* agg1  = ws + off; off += (size_t)N * 128;   // becomes relu'd layer-1 output in place
  float* aexp  = ws + off; off += (size_t)Et * 4;    // reused for layer 2 (needs Et)
  float* asrc1 = ws + off; off += (size_t)N * 4;
  float* adst1 = ws + off; off += (size_t)N * 4;
  float* amax1 = ws + off; off += (size_t)N * 4;
  float* den1  = ws + off; off += (size_t)N * 4;
  float* asrc2 = ws + off; off += (size_t)N;
  float* adst2 = ws + off; off += (size_t)N;
  float* amax2 = ws + off; off += (size_t)N;
  float* den2  = ws + off; off += (size_t)N;
  float* agg2  = ws + off; off += (size_t)N * 32;
  float* h2    = h1;                                  // h1 dead after layer-1 aggregation

  // ---- init accumulators (every call: graph-replay safe) ----
  fill_k<<<1024, 256, 0, stream>>>(amax1, -1e30f, (size_t)N * 4);
  fill_k<<<1024, 256, 0, stream>>>(den1,  0.f,    (size_t)N * 4);
  fill_k<<<4096, 256, 0, stream>>>(agg1,  0.f,    (size_t)N * 128);
  fill_k<<<1024, 256, 0, stream>>>(amax2, -1e30f, (size_t)N);
  fill_k<<<1024, 256, 0, stream>>>(den2,  0.f,    (size_t)N);
  fill_k<<<2048, 256, 0, stream>>>(agg2,  0.f,    (size_t)N * 32);

  const int et4_blocks = (int)((Et * 4 + 255) / 256);
  const int et1_blocks = (int)((Et + 255) / 256);
  const int agg_blocks = (int)((Et + 7) / 8);

  // ---- layer 1 ----
  dim3 g1((N + 15) / 16, 128 / 16);
  gemm_wmma_k<128><<<g1, 32, 0, stream>>>(x, W1, h1, N, 128);
  att_scores_k<4><<<(N * 4 + 255) / 256, 256, 0, stream>>>(
      h1, att_src1, att_dst1, asrc1, adst1, N);
  edge_max_k<4><<<et4_blocks, 256, 0, stream>>>(ei, E, N, asrc1, adst1, amax1);
  edge_exp_k<4><<<et4_blocks, 256, 0, stream>>>(ei, E, N, asrc1, adst1, amax1,
                                                aexp, den1);
  edge_agg_h4_k<<<agg_blocks, 256, 0, stream>>>(ei, E, N, h1, aexp, agg1);
  epilogue_k<<<((long)N * 128 + 255) / 256, 256, 0, stream>>>(
      agg1, den1, bias1, agg1, N, 4, 32);   // in-place -> layer-2 input

  // ---- layer 2 ----
  dim3 g2((N + 15) / 16, 32 / 16);
  gemm_wmma_k<128><<<g2, 32, 0, stream>>>(agg1, W2, h2, N, 32);
  att_scores_k<1><<<(N + 255) / 256, 256, 0, stream>>>(
      h2, att_src2, att_dst2, asrc2, adst2, N);
  edge_max_k<1><<<et1_blocks, 256, 0, stream>>>(ei, E, N, asrc2, adst2, amax2);
  edge_exp_k<1><<<et1_blocks, 256, 0, stream>>>(ei, E, N, asrc2, adst2, amax2,
                                                aexp, den2);
  edge_agg_h1_k<<<agg_blocks, 256, 0, stream>>>(ei, E, N, h2, aexp, agg2);
  epilogue_k<<<((long)N * 32 + 255) / 256, 256, 0, stream>>>(
      agg2, den2, bias2, out, N, 1, 32);
}
